// AttentionHead_38843684225633
// MI455X (gfx1250) — matrix operand order repaired and checked by reading, not compiled, not run on hardware
//
#include <hip/hip_runtime.h>
#include <hip/hip_bf16.h>

typedef __attribute__((ext_vector_type(16))) __bf16 v16bf;
typedef __attribute__((ext_vector_type(8)))  float  v8f;

#define TT 4096
#define DD 1024
#define HH 64
#define BB 4
#define BT (BB*TT)

union BF16x16 {
    v16bf v;
    unsigned int u[8];
    uint4 q[2];
};

// ---- f32 -> bf16 pair pack: prefer hardware v_cvt_pk_bf16_f32 ----
#if __has_builtin(__builtin_amdgcn_cvt_pk_bf16_f32)
__device__ __forceinline__ unsigned int packbf(float lo, float hi) {
    auto r = __builtin_amdgcn_cvt_pk_bf16_f32(lo, hi);
    unsigned int u;
    __builtin_memcpy(&u, &r, sizeof(u));
    return u;
}
#else
__device__ __forceinline__ unsigned int f2bf_sw(float f) {
    unsigned int x = __float_as_uint(f);
    return (x + 0x7FFFu + ((x >> 16) & 1u)) >> 16;   // round-to-nearest-even
}
__device__ __forceinline__ unsigned int packbf(float lo, float hi) {
    return f2bf_sw(lo) | (f2bf_sw(hi) << 16);
}
#endif
__device__ __forceinline__ unsigned short f2bf(float f) {
    return (unsigned short)packbf(f, f);
}

// ---- async global -> LDS copy (16B per lane), tracked by ASYNCcnt ----
__device__ __forceinline__ void async_copy_b128(unsigned int lds_off,
                                                const void* gaddr) {
    asm volatile("global_load_async_to_lds_b128 %0, %1, off"
                 :: "v"(lds_off), "v"(gaddr) : "memory");
}
__device__ __forceinline__ void wait_async_le(int) {}  // (doc helper, unused)
__device__ __forceinline__ void wait_async_2() {
#if __has_builtin(__builtin_amdgcn_s_wait_asynccnt)
    __builtin_amdgcn_s_wait_asynccnt(2);
#else
    asm volatile("s_wait_asynccnt 2" ::: "memory");
#endif
}
__device__ __forceinline__ void wait_async_0() {
#if __has_builtin(__builtin_amdgcn_s_wait_asynccnt)
    __builtin_amdgcn_s_wait_asynccnt(0);
#else
    asm volatile("s_wait_asynccnt 0" ::: "memory");
#endif
}
__device__ __forceinline__ unsigned int lds_addr_of(const void* p) {
    // generic pointer to a __shared__ object: low 32 bits = LDS byte address
    return (unsigned int)(unsigned long long)p;
}

// ---------------------------------------------------------------------------
// Kernel 0: one-shot f32 -> bf16 conversion (x and the three weight matrices).
// ---------------------------------------------------------------------------
__global__ __launch_bounds__(256) void cvt_bf16_kernel(
    const float* __restrict__ src, unsigned short* __restrict__ dst, int n8)
{
    const int i = blockIdx.x * 256 + threadIdx.x;
    if (i >= n8) return;
    const float4* s = (const float4*)(src + (size_t)i * 8);
    float4 a = s[0], b = s[1];
    uint4 o;
    o.x = packbf(a.x, a.y); o.y = packbf(a.z, a.w);
    o.z = packbf(b.x, b.y); o.w = packbf(b.z, b.w);
    *(uint4*)(dst + (size_t)i * 8) = o;
}

// ---------------------------------------------------------------------------
// Kernel 1: QKV projection, all-bf16 operands (pre-converted).
// q' = (x@wq^T + bq)*H^-0.5, k' = x@wk^T + bk, v stored TRANSPOSED [H][B*T].
// One wave = 16 rows x all 64 outputs for q,k,v.  Hot loop: b128 loads + wmma.
// ---------------------------------------------------------------------------
__global__ __launch_bounds__(256) void qkv_kernel(
    const unsigned short* __restrict__ xbf,   // [BT][DD] bf16
    const unsigned short* __restrict__ wqb,   // [HH][DD] bf16
    const unsigned short* __restrict__ wkb,
    const unsigned short* __restrict__ wvb,
    const float* __restrict__ bq, const float* __restrict__ bk,
    const float* __restrict__ bv,
    unsigned short* __restrict__ qo,   // [BT][HH] bf16
    unsigned short* __restrict__ ko,   // [BT][HH] bf16
    unsigned short* __restrict__ vo)   // [HH][BT] bf16 (transposed)
{
    const int lane = threadIdx.x & 31;
    const int wave = threadIdx.x >> 5;
    const int tile = blockIdx.x * 8 + wave;   // 0..1023 row tiles of 16
    const int row0 = tile * 16;
    const int hf   = lane >> 4;               // half-wave select
    const int lm   = lane & 15;

    v8f zero = {};
    v8f acc[3][4];
#pragma unroll
    for (int w = 0; w < 3; ++w)
#pragma unroll
        for (int n = 0; n < 4; ++n) acc[w][n] = zero;

    const unsigned short* wptr[3] = {wqb, wkb, wvb};
    const unsigned int* xrow32 =
        (const unsigned int*)(xbf + (size_t)(row0 + lm) * DD);

    for (int d = 0; d < DD; d += 32) {
        // A operand: lane lm = row; two runs of 8 consecutive bf16
        BF16x16 A;
        A.q[0] = *(const uint4*)(xrow32 + d / 2 + hf * 4);
        A.q[1] = *(const uint4*)(xrow32 + d / 2 + 8 + hf * 4);
#pragma unroll
        for (int w = 0; w < 3; ++w) {
            const unsigned short* W = wptr[w];
#pragma unroll
            for (int n = 0; n < 4; ++n) {
                const int h = n * 16 + lm;
                const unsigned int* w32 =
                    (const unsigned int*)(W + (size_t)h * DD);
                BF16x16 Bm;
                Bm.q[0] = *(const uint4*)(w32 + d / 2 + hf * 4);
                Bm.q[1] = *(const uint4*)(w32 + d / 2 + 8 + hf * 4);
                acc[w][n] = __builtin_amdgcn_wmma_f32_16x16x32_bf16(
                    false, A.v, false, Bm.v, (short)0, acc[w][n], false, false);
            }
        }
    }

    const float qscale = 0.125f;   // 64^-0.5 folded into q
#pragma unroll
    for (int n = 0; n < 4; ++n) {
        const int h = n * 16 + lm;
        const float bqv = bq[h], bkv = bk[h], bvv = bv[h];
#pragma unroll
        for (int r = 0; r < 8; ++r) {
            const int row = row0 + hf * 8 + r;     // C layout: M = 8*half + r
            qo[(size_t)row * HH + h] = f2bf((acc[0][n][r] + bqv) * qscale);
            ko[(size_t)row * HH + h] = f2bf(acc[1][n][r] + bkv);
            vo[(size_t)h * BT + row] = f2bf(acc[2][n][r] + bvv);
        }
    }
}

// ---------------------------------------------------------------------------
// Kernel 2: flash attention with double-buffered async K/V staging in LDS.
// All 8 waves of a block share one batch, so each 32-key chunk of K'/V^T is
// copied to LDS once per block (global_load_async_to_lds_b128, ASYNCcnt),
// overlapping the copy of chunk i+1 with the WMMAs/softmax of chunk i.
// ---------------------------------------------------------------------------
__global__ __launch_bounds__(256) void attn_kernel(
    const unsigned short* __restrict__ qo,
    const unsigned short* __restrict__ ko,
    const unsigned short* __restrict__ vo,
    const int* __restrict__ mask,
    float* __restrict__ out)
{
    __shared__ __align__(16) unsigned short kbuf[2][32 * HH]; // 2 x 4KB
    __shared__ __align__(16) unsigned short vbuf[2][HH * 32]; // 2 x 4KB
    __shared__ __align__(16) unsigned int   p_lds[8][16 * 16]; // per-wave 16x32 bf16

    const int tid   = threadIdx.x;
    const int lane  = tid & 31;
    const int wave  = tid >> 5;
    const int qtile = blockIdx.x * 8 + wave;  // 0..1023
    const int b     = qtile >> 8;             // 256 tiles per batch
    const int t0    = (qtile & 255) * 16;
    const int hf    = lane >> 4;
    const int lm    = lane & 15;

    // staging assignments (16B per thread per matrix)
    const int k_key  = tid >> 3, k_part = tid & 7;   // K: 32 keys x 8 x 16B
    const int v_h    = tid >> 2, v_part = tid & 3;   // V: 64 h   x 4 x 16B
    const unsigned short* kbase = ko + (size_t)b * TT * HH;
    const unsigned short* vbase = vo + (size_t)b * TT;

    // Q: two A-operands (K = h 0..31 and 32..63), lane lm = query row
    BF16x16 Aq[2];
    {
        const unsigned int* q32 =
            (const unsigned int*)(qo + ((size_t)b * TT + t0 + lm) * HH);
#pragma unroll
        for (int ks = 0; ks < 2; ++ks) {
            Aq[ks].q[0] = *(const uint4*)(q32 + ks * 16 + hf * 4);
            Aq[ks].q[1] = *(const uint4*)(q32 + ks * 16 + 8 + hf * 4);
        }
    }

    v8f zero = {};
    v8f O[4];
#pragma unroll
    for (int n = 0; n < 4; ++n) O[n] = zero;
    float m_run[8], l_run[8];
#pragma unroll
    for (int r = 0; r < 8; ++r) { m_run[r] = -3.0e38f; l_run[r] = 0.0f; }

    const int* mrow = mask + b * TT;
    unsigned int* plds = p_lds[wave];

    // ---- prologue: issue async copy of chunk 0 into buffer 0 ----
    async_copy_b128(lds_addr_of(&kbuf[0][k_key * HH + k_part * 8]),
                    kbase + (size_t)k_key * HH + k_part * 8);
    async_copy_b128(lds_addr_of(&vbuf[0][v_h * 32 + v_part * 8]),
                    vbase + (size_t)v_h * BT + v_part * 8);

    const int NCH = TT / 32;   // 128 chunks
    for (int ci = 0; ci < NCH; ++ci) {
        const int s0  = ci * 32;
        const int buf = ci & 1;
        // issue next chunk's copy into the other buffer
        if (ci + 1 < NCH) {
            const int s1 = s0 + 32;
            async_copy_b128(
                lds_addr_of(&kbuf[buf ^ 1][k_key * HH + k_part * 8]),
                kbase + (size_t)(s1 + k_key) * HH + k_part * 8);
            async_copy_b128(
                lds_addr_of(&vbuf[buf ^ 1][v_h * 32 + v_part * 8]),
                vbase + (size_t)v_h * BT + s1 + v_part * 8);
            wait_async_2();        // my share of chunk ci has landed
        } else {
            wait_async_0();
        }
        __syncthreads();           // everyone's share of chunk ci visible

        // ---- S = q'.k'^T from LDS (scale already folded into q') ----
        const unsigned int* k32 = (const unsigned int*)kbuf[buf];
        v8f S[2];
#pragma unroll
        for (int nt = 0; nt < 2; ++nt) {
            const int kl = nt * 16 + lm;             // key within chunk
            v8f c = zero;
#pragma unroll
            for (int ks = 0; ks < 2; ++ks) {
                BF16x16 Bk;
                Bk.q[0] = *(const uint4*)(k32 + kl * 32 + ks * 16 + hf * 4);
                Bk.q[1] = *(const uint4*)(k32 + kl * 32 + ks * 16 + 8 + hf * 4);
                c = __builtin_amdgcn_wmma_f32_16x16x32_bf16(
                    false, Aq[ks].v, false, Bk.v, (short)0, c, false, false);
            }
            const int mk = mrow[s0 + kl];
#pragma unroll
            for (int r = 0; r < 8; ++r)
                c[r] = mk ? c[r] : -1.0e9f;          // match reference masking
            S[nt] = c;
        }

        // ---- online softmax: butterfly over the 16-lane half ----
        float scale[8];
#pragma unroll
        for (int r = 0; r < 8; ++r) {
            float mx = fmaxf(S[0][r], S[1][r]);
            mx = fmaxf(mx, __shfl_xor(mx, 1, 32));
            mx = fmaxf(mx, __shfl_xor(mx, 2, 32));
            mx = fmaxf(mx, __shfl_xor(mx, 4, 32));
            mx = fmaxf(mx, __shfl_xor(mx, 8, 32));
            const float mn = fmaxf(m_run[r], mx);
            scale[r] = __expf(m_run[r] - mn);
            const float p0 = __expf(S[0][r] - mn);
            const float p1 = __expf(S[1][r] - mn);
            S[0][r] = p0; S[1][r] = p1;
            float rs = p0 + p1;
            rs += __shfl_xor(rs, 1, 32);
            rs += __shfl_xor(rs, 2, 32);
            rs += __shfl_xor(rs, 4, 32);
            rs += __shfl_xor(rs, 8, 32);
            l_run[r] = l_run[r] * scale[r] + rs;
            m_run[r] = mn;
        }

        // ---- C-layout f32 P -> A-layout bf16 via LDS (same-wave only) ----
        unsigned short* plds16 = (unsigned short*)plds;
#pragma unroll
        for (int r = 0; r < 8; ++r) {
            const int row = hf * 8 + r;
            plds16[row * 32 + lm]      = f2bf(S[0][r]);
            plds16[row * 32 + 16 + lm] = f2bf(S[1][r]);
        }
        BF16x16 Ap;
        {
            const unsigned int* p32 = plds + lm * 16;   // row lm, 16 uints
            Ap.q[0] = *(const uint4*)(p32 + hf * 4);
            Ap.q[1] = *(const uint4*)(p32 + 8 + hf * 4);
        }

        // ---- O = O*scale + P @ V (V from LDS) ----
        const unsigned int* v32 = (const unsigned int*)vbuf[buf];
#pragma unroll
        for (int n = 0; n < 4; ++n) {
            v8f o = O[n];
#pragma unroll
            for (int r = 0; r < 8; ++r) o[r] *= scale[r];
            const int h = n * 16 + lm;
            BF16x16 Bv;
            Bv.q[0] = *(const uint4*)(v32 + h * 16 + hf * 4);
            Bv.q[1] = *(const uint4*)(v32 + h * 16 + 8 + hf * 4);
            O[n] = __builtin_amdgcn_wmma_f32_16x16x32_bf16(
                false, Ap.v, false, Bv.v, (short)0, o, false, false);
        }
        __syncthreads();           // done reading buf before it is overwritten
    }

    // ---- normalize and store f32 [B,T,H] ----
#pragma unroll
    for (int n = 0; n < 4; ++n) {
        const int h = n * 16 + lm;
#pragma unroll
        for (int r = 0; r < 8; ++r) {
            const int t = t0 + hf * 8 + r;
            out[((size_t)b * TT + t) * HH + h] = O[n][r] / l_run[r];
        }
    }
}

extern "C" void kernel_launch(void* const* d_in, const int* in_sizes, int n_in,
                              void* d_out, int out_size, void* d_ws, size_t ws_size,
                              hipStream_t stream) {
    const float* x    = (const float*)d_in[0];
    const int*   mask = (const int*)d_in[1];
    const float* wq   = (const float*)d_in[2];
    const float* bq   = (const float*)d_in[3];
    const float* wk   = (const float*)d_in[4];
    const float* bk   = (const float*)d_in[5];
    const float* wv   = (const float*)d_in[6];
    const float* bv   = (const float*)d_in[7];
    float* out = (float*)d_out;

    // workspace layout (bytes):
    //   x_bf   [BT][DD] bf16 : 32 MB
    //   wq/wk/wv bf16        : 3 x 128 KB
    //   q' / k' / v^T bf16   : 3 x 2 MB
    char* ws = (char*)d_ws;
    unsigned short* xbf = (unsigned short*)(ws);
    size_t off = (size_t)BT * DD * 2;
    unsigned short* wqb = (unsigned short*)(ws + off); off += (size_t)HH * DD * 2;
    unsigned short* wkb = (unsigned short*)(ws + off); off += (size_t)HH * DD * 2;
    unsigned short* wvb = (unsigned short*)(ws + off); off += (size_t)HH * DD * 2;
    unsigned short* qo  = (unsigned short*)(ws + off); off += (size_t)BT * HH * 2;
    unsigned short* ko  = (unsigned short*)(ws + off); off += (size_t)BT * HH * 2;
    unsigned short* vo  = (unsigned short*)(ws + off);

    const int n8x = BT * DD / 8;
    const int n8w = HH * DD / 8;
    cvt_bf16_kernel<<<(n8x + 255) / 256, 256, 0, stream>>>(x,  xbf, n8x);
    cvt_bf16_kernel<<<(n8w + 255) / 256, 256, 0, stream>>>(wq, wqb, n8w);
    cvt_bf16_kernel<<<(n8w + 255) / 256, 256, 0, stream>>>(wk, wkb, n8w);
    cvt_bf16_kernel<<<(n8w + 255) / 256, 256, 0, stream>>>(wv, wvb, n8w);

    qkv_kernel<<<128, 256, 0, stream>>>(xbf, wqb, wkb, wvb, bq, bk, bv,
                                        qo, ko, vo);
    attn_kernel<<<128, 256, 0, stream>>>(qo, ko, vo, mask, out);
}
